// MultiHeadAttention_59502476919427
// MI455X (gfx1250) — compile-verified, matrix operand order checked by reading, not run on hardware
//
#include <hip/hip_runtime.h>
#include <hip/hip_bf16.h>

#define D_MODEL 1024
#define DK 64
#define NH 16
#define BB 4
#define SS 2048

typedef __attribute__((ext_vector_type(16))) __bf16 v16bf;
typedef __attribute__((ext_vector_type(8)))  float  v8f;
typedef __attribute__((ext_vector_type(4)))  float  v4f;
typedef __attribute__((ext_vector_type(8)))  unsigned short us8;

struct US8x2 { us8 lo; us8 hi; };

static __device__ __forceinline__ unsigned short f2bf(float f) {
  unsigned int u = __float_as_uint(f);
  u += 0x7FFFu + ((u >> 16) & 1u);      // round-to-nearest-even
  return (unsigned short)(u >> 16);
}

// A/B fragment from bf16 row (row-major, contiguous K). Per ISA 16-bit A layout:
// lane-half hi=0 holds K = [0..7] and [16..23]; hi=1 holds K = [8..15] and [24..31].
static __device__ __forceinline__ v16bf frag_bf16(const unsigned short* p, int hi) {
  US8x2 t;
  t.lo = *reinterpret_cast<const us8*>(p + hi * 8);
  t.hi = *reinterpret_cast<const us8*>(p + 16 + hi * 8);
  return __builtin_bit_cast(v16bf, t);
}

static __device__ __forceinline__ v8f wmma_bf16(v16bf a, v16bf b, v8f c) {
  return __builtin_amdgcn_wmma_f32_16x16x32_bf16(false, a, false, b, (short)0, c,
                                                 false, false);
}

// ---- one-pass fp32 -> bf16 activation conversion (8 elts/thread, b128 I/O) ----

__global__ void k_cvt_bf16(const float* __restrict__ in, unsigned short* __restrict__ out,
                           int n /* multiple of 8 */) {
  int i = (blockIdx.x * blockDim.x + threadIdx.x) * 8;
  if (i >= n) return;
  v4f a = *reinterpret_cast<const v4f*>(in + i);
  v4f b = *reinterpret_cast<const v4f*>(in + i + 4);
  us8 o;
#pragma unroll
  for (int j = 0; j < 4; ++j) { o[j] = f2bf(a[j]); o[4 + j] = f2bf(b[j]); }
  *reinterpret_cast<us8*>(out + i) = o;
}

// ---- weight pre-transpose (fp32 -> bf16, contraction dim made contiguous) ----

__global__ void k_tr_headw(const float* __restrict__ in, unsigned short* __restrict__ out) {
  int idx = blockIdx.x * blockDim.x + threadIdx.x;       // NH*DK*D_MODEL = 1M
  if (idx >= NH * DK * D_MODEL) return;
  int d = idx & (D_MODEL - 1);
  int n = (idx >> 10) & (DK - 1);
  int h = idx >> 16;
  out[idx] = f2bf(in[((size_t)h * D_MODEL + d) * DK + n]);   // Wt[h][n][d] = W[h][d][n]
}

__global__ void k_tr_wo(const float* __restrict__ in, unsigned short* __restrict__ out) {
  int idx = blockIdx.x * blockDim.x + threadIdx.x;       // 1024*1024
  if (idx >= D_MODEL * D_MODEL) return;
  int kk = idx & (D_MODEL - 1);
  int n  = idx >> 10;
  out[idx] = f2bf(in[(size_t)kk * D_MODEL + n]);          // Wot[n][k] = Wo[k][n]
}

// ---- fused QKV projection: (B*S,1024) x per-head (1024,64), bf16 WMMA ----
// grid (512 row-tiles, 16 heads, 3={q,k,v}); 4 waves/block, one 16x16 tile each.
// Inner loop per wave: 4x global_load_b128 + 1x v_wmma (no conversion VALU).

__global__ __launch_bounds__(128) void k_qkv_proj(
    const unsigned short* __restrict__ qb, const unsigned short* __restrict__ kb,
    const unsigned short* __restrict__ vb,
    const unsigned short* __restrict__ Wqt, const unsigned short* __restrict__ Wkt,
    const unsigned short* __restrict__ Wvt,
    const float* __restrict__ bq, const float* __restrict__ bk, const float* __restrict__ bv,
    unsigned short* __restrict__ qh, unsigned short* __restrict__ kh,
    unsigned short* __restrict__ vt)
{
  const int rt    = blockIdx.x;           // row tile over B*S (512)
  const int h     = blockIdx.y;           // head
  const int which = blockIdx.z;           // 0=q 1=k 2=v
  const int w     = threadIdx.x >> 5;     // n-subtile 0..3
  const int lane  = threadIdx.x & 31;
  const int m     = lane & 15;
  const int hi    = lane >> 4;

  const unsigned short* in   = (which == 0) ? qb  : (which == 1) ? kb  : vb;
  const unsigned short* Wt   = (which == 0) ? Wqt : (which == 1) ? Wkt : Wvt;
  const float* bias          = (which == 0) ? bq  : (which == 1) ? bk  : bv;

  const int ncol = w * 16 + m;
  const unsigned short* arow = in + (size_t)(rt * 16 + m) * D_MODEL;
  const unsigned short* brow = Wt + ((size_t)h * DK + ncol) * D_MODEL;

  v8f acc = {};
#pragma unroll 4
  for (int kt = 0; kt < D_MODEL / 32; ++kt) {
    v16bf a = frag_bf16(arow + kt * 32, hi);
    v16bf b = frag_bf16(brow + kt * 32, hi);
    acc = wmma_bf16(a, b, acc);
  }
  const float bsv = bias[h * DK + ncol];

#pragma unroll
  for (int r = 0; r < 8; ++r) {
    int row = rt * 16 + r + 8 * hi;               // global row over B*S
    int b_i = row / SS, s_i = row % SS;
    unsigned short o16 = f2bf(acc[r] + bsv);
    if (which == 2) {
      // V stored transposed: vt[b][h][dv][s] so PV B-fragments are contiguous in s.
      vt[(((size_t)b_i * NH + h) * DK + ncol) * SS + s_i] = o16;
    } else {
      unsigned short* outp = (which == 0) ? qh : kh;
      outp[(((size_t)b_i * NH + h) * SS + s_i) * DK + ncol] = o16;
    }
  }
}

// ---- flash attention per (b*h, 16-query tile); 1 wave per block ----

__global__ __launch_bounds__(32) void k_flash(
    const unsigned short* __restrict__ qh, const unsigned short* __restrict__ kh,
    const unsigned short* __restrict__ vt, unsigned short* __restrict__ ctx)
{
  __shared__ __align__(16) unsigned short plds[16 * 32];
  const int bh   = blockIdx.x;   // 0..63 = b*16+h
  const int qt   = blockIdx.y;   // 0..127 query tile
  const int lane = threadIdx.x & 31;
  const int m    = lane & 15;
  const int hi   = lane >> 4;

  const unsigned short* qbase = qh + ((size_t)bh * SS + qt * 16 + m) * DK;
  const v16bf qa0 = frag_bf16(qbase, hi);        // dk 0..31
  const v16bf qa1 = frag_bf16(qbase + 32, hi);   // dk 32..63

  v8f acc[4] = {v8f{}, v8f{}, v8f{}, v8f{}};
  float mrow[8], lrow[8];
#pragma unroll
  for (int r = 0; r < 8; ++r) { mrow[r] = -1e30f; lrow[r] = 0.f; }

  const unsigned short* kbh = kh + (size_t)bh * SS * DK;
  const unsigned short* vbh = vt + (size_t)bh * DK * SS;

  for (int j = 0; j < SS / 32; ++j) {
    // scores: two 16x16 tiles over this 32-key chunk, contraction dk=64
    const unsigned short* kr0 = kbh + (size_t)(j * 32 + m) * DK;
    const unsigned short* kr1 = kbh + (size_t)(j * 32 + 16 + m) * DK;
    v8f s0 = {}, s1 = {};
    s0 = wmma_bf16(qa0, frag_bf16(kr0, hi), s0);
    s0 = wmma_bf16(qa1, frag_bf16(kr0 + 32, hi), s0);
    s1 = wmma_bf16(qa0, frag_bf16(kr1, hi), s1);
    s1 = wmma_bf16(qa1, frag_bf16(kr1 + 32, hi), s1);

    // online softmax (row stats per VGPR slot r; 16-lane xor reductions stay
    // inside each lane half, matching the C/D M = r + 8*hi mapping)
    float p0[8], p1[8];
#pragma unroll
    for (int r = 0; r < 8; ++r) {
      float a = s0[r] * 0.125f, b = s1[r] * 0.125f;   // 1/sqrt(64)
      float mx = fmaxf(a, b);
#pragma unroll
      for (int o = 1; o < 16; o <<= 1) mx = fmaxf(mx, __shfl_xor(mx, o, 32));
      float mnew = fmaxf(mrow[r], mx);
      float fac  = __expf(mrow[r] - mnew);
      float e0 = __expf(a - mnew), e1 = __expf(b - mnew);
      float sum = e0 + e1;
#pragma unroll
      for (int o = 1; o < 16; o <<= 1) sum += __shfl_xor(sum, o, 32);
      lrow[r] = lrow[r] * fac + sum;
      mrow[r] = mnew;
#pragma unroll
      for (int n = 0; n < 4; ++n) acc[n][r] *= fac;
      p0[r] = e0; p1[r] = e1;
    }

    // C/D-layout P -> LDS (row-major 16x32 bf16) -> reload as A fragment
#pragma unroll
    for (int r = 0; r < 8; ++r) {
      int row = r + 8 * hi;
      plds[row * 32 + m]      = f2bf(p0[r]);
      plds[row * 32 + 16 + m] = f2bf(p1[r]);
    }
    __syncthreads();                       // 1-wave block: compiler fence + waits
    v16bf pa = frag_bf16(&plds[m * 32], hi);
#pragma unroll
    for (int n = 0; n < 4; ++n) {
      const unsigned short* vr = vbh + (size_t)(n * 16 + m) * SS + j * 32;
      acc[n] = wmma_bf16(pa, frag_bf16(vr, hi), acc[n]);
    }
    __syncthreads();
  }

  // normalize and store ctx as bf16 in (B, S, H*dv) row-major
  const int b_i = bh >> 4, h_i = bh & 15;
#pragma unroll
  for (int r = 0; r < 8; ++r) {
    float inv = 1.0f / lrow[r];
    int srow = qt * 16 + r + 8 * hi;
    size_t o = ((size_t)b_i * SS + srow) * (NH * DK) + (size_t)h_i * DK;
#pragma unroll
    for (int n = 0; n < 4; ++n)
      ctx[o + n * 16 + m] = f2bf(acc[n][r] * inv);
  }
}

// ---- output projection: (8192,1024) x (1024,1024) + bo, fp32 out ----

__global__ __launch_bounds__(128) void k_outproj(
    const unsigned short* __restrict__ ctx, const unsigned short* __restrict__ Wot,
    const float* __restrict__ bo, float* __restrict__ out)
{
  const int rt   = blockIdx.x;      // 512 row tiles
  const int cg   = blockIdx.y;      // 16 column groups of 64
  const int w    = threadIdx.x >> 5;
  const int lane = threadIdx.x & 31;
  const int m    = lane & 15;
  const int hi   = lane >> 4;
  const int ncol = cg * 64 + w * 16 + m;

  const unsigned short* arow = ctx + (size_t)(rt * 16 + m) * D_MODEL;
  const unsigned short* brow = Wot + (size_t)ncol * D_MODEL;

  v8f acc = {};
#pragma unroll 4
  for (int kt = 0; kt < D_MODEL / 32; ++kt)
    acc = wmma_bf16(frag_bf16(arow + kt * 32, hi), frag_bf16(brow + kt * 32, hi), acc);

  const float bias = bo[ncol];
#pragma unroll
  for (int r = 0; r < 8; ++r) {
    int row = rt * 16 + r + 8 * hi;
    out[(size_t)row * D_MODEL + ncol] = acc[r] + bias;
  }
}

extern "C" void kernel_launch(void* const* d_in, const int* in_sizes, int n_in,
                              void* d_out, int out_size, void* d_ws, size_t ws_size,
                              hipStream_t stream) {
  const float* q  = (const float*)d_in[0];
  const float* k  = (const float*)d_in[1];
  const float* v  = (const float*)d_in[2];
  const float* Wq = (const float*)d_in[3];
  const float* bq = (const float*)d_in[4];
  const float* Wk = (const float*)d_in[5];
  const float* bk = (const float*)d_in[6];
  const float* Wv = (const float*)d_in[7];
  const float* bv = (const float*)d_in[8];
  const float* Wo = (const float*)d_in[9];
  const float* bo = (const float*)d_in[10];
  float* out = (float*)d_out;

  // workspace partition (bf16 buffers)
  char* ws = (char*)d_ws;
  const size_t ACT  = (size_t)BB * SS * D_MODEL * 2;        // 16.8 MB each
  const size_t WSZ  = (size_t)NH * DK * D_MODEL * 2;        // 2 MB each
  const size_t HSZ  = (size_t)BB * NH * SS * DK * 2;        // 16.8 MB each
  unsigned short* qcv = (unsigned short*)(ws);               ws += ACT;
  unsigned short* kcv = (unsigned short*)(ws);               ws += ACT;
  unsigned short* vcv = (unsigned short*)(ws);               ws += ACT;
  unsigned short* Wqt = (unsigned short*)(ws);               ws += WSZ;
  unsigned short* Wkt = (unsigned short*)(ws);               ws += WSZ;
  unsigned short* Wvt = (unsigned short*)(ws);               ws += WSZ;
  unsigned short* Wot = (unsigned short*)(ws);               ws += (size_t)D_MODEL * D_MODEL * 2;
  unsigned short* qhb = (unsigned short*)(ws);               ws += HSZ;
  unsigned short* khb = (unsigned short*)(ws);               ws += HSZ;
  unsigned short* vtb = (unsigned short*)(ws);               ws += HSZ;
  unsigned short* ctx = (unsigned short*)(ws);               ws += HSZ;
  (void)in_sizes; (void)n_in; (void)out_size; (void)ws_size;

  const int tw = 256;
  const int nact = BB * SS * D_MODEL;
  k_cvt_bf16<<<(nact / 8 + tw - 1) / tw, tw, 0, stream>>>(q, qcv, nact);
  k_cvt_bf16<<<(nact / 8 + tw - 1) / tw, tw, 0, stream>>>(k, kcv, nact);
  k_cvt_bf16<<<(nact / 8 + tw - 1) / tw, tw, 0, stream>>>(v, vcv, nact);

  k_tr_headw<<<(NH * DK * D_MODEL + tw - 1) / tw, tw, 0, stream>>>(Wq, Wqt);
  k_tr_headw<<<(NH * DK * D_MODEL + tw - 1) / tw, tw, 0, stream>>>(Wk, Wkt);
  k_tr_headw<<<(NH * DK * D_MODEL + tw - 1) / tw, tw, 0, stream>>>(Wv, Wvt);
  k_tr_wo  <<<(D_MODEL * D_MODEL + tw - 1) / tw, tw, 0, stream>>>(Wo, Wot);

  k_qkv_proj<<<dim3(BB * SS / 16, NH, 3), 128, 0, stream>>>(
      qcv, kcv, vcv, Wqt, Wkt, Wvt, bq, bk, bv, qhb, khb, vtb);

  k_flash<<<dim3(BB * NH, SS / 16), 32, 0, stream>>>(qhb, khb, vtb, ctx);

  k_outproj<<<dim3(BB * SS / 16, NH), 128, 0, stream>>>(ctx, Wot, bo, out);
}
